// CrossAttentionLayer_11218454577243
// MI455X (gfx1250) — matrix operand order repaired and checked by reading, not compiled, not run on hardware
//
#include <hip/hip_runtime.h>

// ---------- types ----------
typedef __attribute__((ext_vector_type(16))) __bf16 v16bf;
typedef __attribute__((ext_vector_type(8)))  float  v8f;

union Frag16 {
    uint4  q[2];
    v16bf  v;
    __bf16 h[16];
};

__device__ __forceinline__ v8f v8f_zero() {
    v8f r;
#pragma unroll
    for (int i = 0; i < 8; ++i) r[i] = 0.0f;
    return r;
}

__device__ __forceinline__ v8f wmma_bf16(v16bf a, v16bf b, v8f c) {
    // D = A(16x32 bf16) * B(32x16 bf16) + C(16x16 f32)
    return __builtin_amdgcn_wmma_f32_16x16x32_bf16(
        /*neg_a=*/false, a, /*neg_b=*/false, b,
        /*c_mod=*/(short)0, c, /*reuse_a=*/false, /*reuse_b=*/false);
}

// ---------- CDNA5 async global->LDS copy (ASYNCcnt-tracked DMA path) ----------
// Per-lane: LDS[vdst] = MEM[vaddr], 16 bytes. Low 32 bits of a generic LDS
// pointer are the LDS byte offset (LDS aperture: addr[31:0] = offset).
__device__ __forceinline__ void async_load_b128(const void* gsrc, void* lds_dst) {
    unsigned lds = (unsigned)(uintptr_t)lds_dst;
    unsigned long long ga = (unsigned long long)(uintptr_t)gsrc;
    asm volatile("global_load_async_to_lds_b128 %0, %1, off"
                 :: "v"(lds), "v"(ga) : "memory");
}
__device__ __forceinline__ void wait_async0() {
#if __has_builtin(__builtin_amdgcn_s_wait_asynccnt)
    __builtin_amdgcn_s_wait_asynccnt(0);
#else
    asm volatile("s_wait_asynccnt 0x0" ::: "memory");
#endif
}

// ---------- DPP 16-lane row reductions (no LDS, no DScnt waits) ----------
// quad_perm(1,0,3,2)=0xB1, quad_perm(2,3,0,1)=0x4E, row_ror:4=0x124, row_ror:8=0x128.
// After the 4 steps every lane in each 16-lane row holds the full row reduction.
template <int CTRL>
__device__ __forceinline__ float dpp_mov_f32(float x) {
    int xi = __float_as_int(x);
    int r  = __builtin_amdgcn_update_dpp(xi, xi, CTRL, 0xF, 0xF, false);
    return __int_as_float(r);
}
__device__ __forceinline__ float row_max16(float x) {
    x = fmaxf(x, dpp_mov_f32<0xB1>(x));
    x = fmaxf(x, dpp_mov_f32<0x4E>(x));
    x = fmaxf(x, dpp_mov_f32<0x124>(x));
    x = fmaxf(x, dpp_mov_f32<0x128>(x));
    return x;
}
__device__ __forceinline__ float row_sum16(float x) {
    x += dpp_mov_f32<0xB1>(x);
    x += dpp_mov_f32<0x4E>(x);
    x += dpp_mov_f32<0x124>(x);
    x += dpp_mov_f32<0x128>(x);
    return x;
}

// ---------- fp32 -> bf16 conversion (vectorized x4) ----------
__global__ __launch_bounds__(256) void cvt_f32_bf16_x4(
    const float* __restrict__ in, __bf16* __restrict__ out, int n4)
{
    int i = blockIdx.x * blockDim.x + threadIdx.x;
    if (i >= n4) return;
    float4 f = reinterpret_cast<const float4*>(in)[i];
    union { __bf16 h[4]; uint2 u; } p;
    p.h[0] = (__bf16)f.x; p.h[1] = (__bf16)f.y;
    p.h[2] = (__bf16)f.z; p.h[3] = (__bf16)f.w;
    reinterpret_cast<uint2*>(out)[i] = p.u;
}

// ---------- WMMA GEMM: C(MxN) = A(MxK, bf16) * W(KxN, bf16) + bias ----------
// Block: 256 thr / 8 waves, tile 128x128x32. Waves 4(m) x 2(n); each wave 32x64.
template <bool OUT_BF16>
__global__ __launch_bounds__(256) void gemm_bias_wmma(
    const __bf16* __restrict__ A, const __bf16* __restrict__ W,
    const float* __restrict__ bias, void* __restrict__ Cout,
    int M, int N, int K)
{
    __shared__ __bf16 Al[128 * 40];   // [row][k], stride 40 halves (80B)
    __shared__ __bf16 Bt[128 * 40];   // [col][k] (W transposed), stride 40

    const int m0  = blockIdx.y * 128;
    const int n0  = blockIdx.x * 128;
    const int tid = threadIdx.x;
    const int wid = tid >> 5, lane = tid & 31;
    const int l15 = lane & 15, hi = lane >> 4;
    const int wm = wid >> 1;          // 0..3
    const int wn = wid & 1;           // 0..1
    const int kb = hi * 8;            // A-fragment half-offset per ISA layout

    v8f acc[2][4];
#pragma unroll
    for (int i = 0; i < 2; ++i)
#pragma unroll
        for (int j = 0; j < 4; ++j) acc[i][j] = v8f_zero();

    const int a_row = tid >> 1, a_c0 = (tid & 1) * 16;
    const int b_kr  = tid >> 3, b_c0 = (tid & 7) * 16;

    for (int k0 = 0; k0 < K; k0 += 32) {
        __syncthreads();
        {   // A tile: 128x32, row-major copy via async DMA (global -> LDS)
            const __bf16* src = A + (size_t)(m0 + a_row) * K + k0 + a_c0;
            __bf16* dst = Al + a_row * 40 + a_c0;
            async_load_b128(src,     dst);
            async_load_b128(src + 8, dst + 8);
        }
        {   // W tile: 32x128, store transposed -> Bt[col][k]
            const __bf16* src = W + (size_t)(k0 + b_kr) * N + n0 + b_c0;
            Frag16 t;
            t.q[0] = reinterpret_cast<const uint4*>(src)[0];
            t.q[1] = reinterpret_cast<const uint4*>(src)[1];
#pragma unroll
            for (int j = 0; j < 16; ++j)
                Bt[(b_c0 + j) * 40 + b_kr] = t.h[j];
        }
        wait_async0();
        __syncthreads();

        v16bf af[2], bfr[4];
#pragma unroll
        for (int tm = 0; tm < 2; ++tm) {
            const __bf16* p = Al + (wm * 32 + tm * 16 + l15) * 40;
            Frag16 u;
            u.q[0] = *reinterpret_cast<const uint4*>(p + kb);       // K kb..kb+7
            u.q[1] = *reinterpret_cast<const uint4*>(p + kb + 16);  // K kb+16..kb+23
            af[tm] = u.v;
        }
#pragma unroll
        for (int tn = 0; tn < 4; ++tn) {
            const __bf16* p = Bt + (wn * 64 + tn * 16 + l15) * 40 + hi * 8;
            Frag16 u;
            u.q[0] = *reinterpret_cast<const uint4*>(p);
            u.q[1] = *reinterpret_cast<const uint4*>(p + 16);
            bfr[tn] = u.v;
        }
#pragma unroll
        for (int tm = 0; tm < 2; ++tm)
#pragma unroll
            for (int tn = 0; tn < 4; ++tn)
                acc[tm][tn] = wmma_bf16(af[tm], bfr[tn], acc[tm][tn]);
    }

    float bv[4];
#pragma unroll
    for (int tn = 0; tn < 4; ++tn)
        bv[tn] = bias[n0 + wn * 64 + tn * 16 + l15];

#pragma unroll
    for (int tm = 0; tm < 2; ++tm)
#pragma unroll
        for (int tn = 0; tn < 4; ++tn) {
            const int n = n0 + wn * 64 + tn * 16 + l15;
#pragma unroll
            for (int v = 0; v < 8; ++v) {
                const int m = m0 + wm * 32 + tm * 16 + v + 8 * hi;
                float val = acc[tm][tn][v] + bv[tn];
                if (OUT_BF16)
                    reinterpret_cast<__bf16*>(Cout)[(size_t)m * N + n] = (__bf16)val;
                else
                    reinterpret_cast<float*>(Cout)[(size_t)m * N + n] = val;
            }
        }
}

// ---------- Flash attention, per (b, h); Q/K/V bf16 with row stride 1024 ----------
// Block: 128 thr / 4 waves, 64 query rows per block, KV blocks of 64 (16 WMMA/iter).
__global__ __launch_bounds__(128) void attn_fwd_wmma(
    const __bf16* __restrict__ Q, const __bf16* __restrict__ Kb,
    const __bf16* __restrict__ Vb, __bf16* __restrict__ O,
    int N1, int N2)
{
    constexpr int HD = 64, E = 1024;
    const float SCALE = 0.25f;   // NUM_HEADS^-0.5

    __shared__ __bf16 Kl[64 * 72];      // [key][d], stride 72 (36-bank step)
    __shared__ __bf16 Vl[64 * 72];      // [d][key] transposed, stride 72
    __shared__ __bf16 Pl[4][16 * 72];   // per-wave P staging (C->A layout swap)

    const int bh = blockIdx.y;
    const int b  = bh >> 4, h = bh & 15;
    const int q0 = blockIdx.x * 64;
    const int tid = threadIdx.x;
    const int wid = tid >> 5, lane = tid & 31;
    const int l15 = lane & 15, hi = lane >> 4;
    const int kb  = hi * 8;

    // Q A-fragments (row = l15 within wave tile), 2 chunks of K=32 over head dim
    v16bf aq[2];
    {
        const __bf16* qp = Q + (size_t)(b * N1 + q0 + wid * 16 + l15) * E + h * HD;
#pragma unroll
        for (int c = 0; c < 2; ++c) {
            Frag16 u;
            u.q[0] = *reinterpret_cast<const uint4*>(qp + c * 32 + kb);
            u.q[1] = *reinterpret_cast<const uint4*>(qp + c * 32 + kb + 16);
            aq[c] = u.v;
        }
    }

    v8f of[4];
#pragma unroll
    for (int n = 0; n < 4; ++n) of[n] = v8f_zero();
    float m_i[8], l_i[8];
#pragma unroll
    for (int v = 0; v < 8; ++v) { m_i[v] = -INFINITY; l_i[v] = 0.0f; }

    const int ld_row = tid >> 1;          // 0..63 (key row)
    const int ld_c0  = (tid & 1) * 32;    // 0 or 32 (head-dim chunk)

    for (int j0 = 0; j0 < N2; j0 += 64) {
        __syncthreads();
        {   // K block, row-major copy via async DMA (global -> LDS)
            const __bf16* src = Kb + (size_t)(b * N2 + j0 + ld_row) * E + h * HD + ld_c0;
            __bf16* dst = Kl + ld_row * 72 + ld_c0;
            async_load_b128(src,     dst);
            async_load_b128(src + 8, dst + 8);
        }
        {   // V block, transposed -> Vl[d][key]
            const __bf16* src = Vb + (size_t)(b * N2 + j0 + ld_row) * E + h * HD + ld_c0;
            Frag16 t0, t1;
            t0.q[0] = reinterpret_cast<const uint4*>(src)[0];
            t0.q[1] = reinterpret_cast<const uint4*>(src)[1];
            t1.q[0] = reinterpret_cast<const uint4*>(src)[2];
            t1.q[1] = reinterpret_cast<const uint4*>(src)[3];
#pragma unroll
            for (int j = 0; j < 16; ++j)
                Vl[(ld_c0 + j) * 72 + ld_row] = t0.h[j];
#pragma unroll
            for (int j = 0; j < 16; ++j)
                Vl[(ld_c0 + 16 + j) * 72 + ld_row] = t1.h[j];
        }
        wait_async0();
        __syncthreads();

        // S = Q * K^T  (per wave: 16 q-rows x 64 keys, 4 WMMA-N tiles, 2 K-chunks)
        v8f s[4];
#pragma unroll
        for (int kt = 0; kt < 4; ++kt) {
            const __bf16* kp = Kl + (kt * 16 + l15) * 72 + hi * 8;
            v8f c = v8f_zero();
#pragma unroll
            for (int dc = 0; dc < 2; ++dc) {
                Frag16 u;
                u.q[0] = *reinterpret_cast<const uint4*>(kp + dc * 32);
                u.q[1] = *reinterpret_cast<const uint4*>(kp + dc * 32 + 16);
                c = wmma_bf16(aq[dc], u.v, c);
            }
            s[kt] = c;
        }

        // Online softmax: rows live at (VGPR v, half hi); DPP 16-lane reductions
#pragma unroll
        for (int v = 0; v < 8; ++v) {
            float sv[4];
#pragma unroll
            for (int t = 0; t < 4; ++t) sv[t] = s[t][v] * SCALE;
            float mx = fmaxf(fmaxf(sv[0], sv[1]), fmaxf(sv[2], sv[3]));
            mx = row_max16(mx);
            float mnew = fmaxf(m_i[v], mx);
            float sc   = __expf(m_i[v] - mnew);
            m_i[v] = mnew;
            float rs = 0.0f;
#pragma unroll
            for (int t = 0; t < 4; ++t) {
                float p = __expf(sv[t] - mnew);
                s[t][v] = p;
                rs += p;
            }
            rs = row_sum16(rs);
            l_i[v] = l_i[v] * sc + rs;
#pragma unroll
            for (int n = 0; n < 4; ++n) of[n][v] *= sc;
        }

        // P: C-layout -> LDS -> A-layout (wave-private; DS ops in-order per wave)
        __bf16* pl = &Pl[wid][0];
#pragma unroll
        for (int kt = 0; kt < 4; ++kt)
#pragma unroll
            for (int v = 0; v < 8; ++v)
                pl[(v + 8 * hi) * 72 + kt * 16 + l15] = (__bf16)s[kt][v];
        v16bf pa[2];
#pragma unroll
        for (int c = 0; c < 2; ++c) {
            const __bf16* pr = pl + l15 * 72 + c * 32;
            Frag16 u;
            u.q[0] = *reinterpret_cast<const uint4*>(pr + kb);
            u.q[1] = *reinterpret_cast<const uint4*>(pr + kb + 16);
            pa[c] = u.v;
        }

        // O += P * V   (contraction over 64 keys = 2 chunks, 4 head-dim tiles)
#pragma unroll
        for (int n = 0; n < 4; ++n) {
#pragma unroll
            for (int c = 0; c < 2; ++c) {
                const __bf16* vp = Vl + (n * 16 + l15) * 72 + c * 32 + hi * 8;
                Frag16 u;
                u.q[0] = *reinterpret_cast<const uint4*>(vp);
                u.q[1] = *reinterpret_cast<const uint4*>(vp + 16);
                of[n] = wmma_bf16(pa[c], u.v, of[n]);
            }
        }
    }

    // Normalize and store O (bf16, head h slice of (B*N1, 1024))
#pragma unroll
    for (int v = 0; v < 8; ++v) {
        float inv = 1.0f / l_i[v];
        const size_t row = (size_t)(b * N1 + q0 + wid * 16 + v + 8 * hi);
#pragma unroll
        for (int n = 0; n < 4; ++n)
            O[row * E + h * HD + n * 16 + l15] = (__bf16)(of[n][v] * inv);
    }
}

// ---------- host launch ----------
extern "C" void kernel_launch(void* const* d_in, const int* in_sizes, int n_in,
                              void* d_out, int out_size, void* d_ws, size_t ws_size,
                              hipStream_t stream) {
    const float* query = (const float*)d_in[0];
    const float* key_  = (const float*)d_in[1];
    const float* value = (const float*)d_in[2];
    const float* Wq = (const float*)d_in[3]; const float* bq = (const float*)d_in[4];
    const float* Wk = (const float*)d_in[5]; const float* bk = (const float*)d_in[6];
    const float* Wv = (const float*)d_in[7]; const float* bv = (const float*)d_in[8];
    const float* Wo = (const float*)d_in[9]; const float* bo = (const float*)d_in[10];

    const int N1 = 2048, N2 = 2048, E = 1024, KV = 768, H = 16;
    const int B = in_sizes[0] / (N1 * E);
    const int M = B * N1;

    __bf16* w = (__bf16*)d_ws;
    __bf16* qbf = w; w += (size_t)M * E;
    __bf16* kbf = w; w += (size_t)M * KV;
    __bf16* vbf = w; w += (size_t)M * KV;
    __bf16* wqb = w; w += (size_t)E * E;
    __bf16* wkb = w; w += (size_t)KV * E;
    __bf16* wvb = w; w += (size_t)KV * E;
    __bf16* wob = w; w += (size_t)E * E;
    __bf16* Qp  = w; w += (size_t)M * E;
    __bf16* Kp  = w; w += (size_t)M * E;
    __bf16* Vp  = w; w += (size_t)M * E;
    __bf16* Op  = w; w += (size_t)M * E;

    auto cvt = [&](const float* src, __bf16* dst, size_t n) {
        int n4 = (int)(n / 4);
        cvt_f32_bf16_x4<<<(n4 + 255) / 256, 256, 0, stream>>>(src, dst, n4);
    };
    cvt(query, qbf, (size_t)M * E);
    cvt(key_,  kbf, (size_t)M * KV);
    cvt(value, vbf, (size_t)M * KV);
    cvt(Wq, wqb, (size_t)E * E);
    cvt(Wk, wkb, (size_t)KV * E);
    cvt(Wv, wvb, (size_t)KV * E);
    cvt(Wo, wob, (size_t)E * E);

    dim3 blk(256);
    dim3 g1(E / 128, M / 128);
    gemm_bias_wmma<true><<<g1, blk, 0, stream>>>(qbf, wqb, bq, Qp, M, E, E);
    gemm_bias_wmma<true><<<g1, blk, 0, stream>>>(kbf, wkb, bk, Kp, M, E, KV);
    gemm_bias_wmma<true><<<g1, blk, 0, stream>>>(vbf, wvb, bv, Vp, M, E, KV);

    dim3 ga(N1 / 64, B * H);
    attn_fwd_wmma<<<ga, 128, 0, stream>>>(Qp, Kp, Vp, Op, N1, N2);

    gemm_bias_wmma<false><<<g1, blk, 0, stream>>>(Op, wob, bo, d_out, M, E, E);
}
